// RouteClassifierGNN_64476049047623
// MI455X (gfx1250) — matrix operand order repaired and checked by reading, not compiled, not run on hardware
//
#include <hip/hip_runtime.h>
#include <hip/hip_bf16.h>

// MI455X / gfx1250 implementation.
// Dense math runs on v_wmma_f32_16x16x32_f16. Weights are pre-packed into
// WMMA B-fragment lane order so B operands are contiguous 32B/lane loads.
// Node state (h, cell) stays resident in f32 (12.8MB each -> lives in L2);
// f16 shadow copies feed the WMMA A-tiles. All tile staging moves 16B chunks
// (global_load_b128 / ds_store_b128); the edge scatter uses float4 loads with
// 4 f32 atomics per thread.

typedef __attribute__((ext_vector_type(16))) _Float16 v16h;
typedef __attribute__((ext_vector_type(8)))  _Float16 v8h;
typedef __attribute__((ext_vector_type(8)))  float    v8f;
typedef __attribute__((ext_vector_type(4)))  float    v4f;

#define HD 64

__device__ __forceinline__ float sigmoidf_(float x) { return 1.0f / (1.0f + __expf(-x)); }

// A fragment (16x32 f16) from an LDS tile with row stride ldk halves.
// Contiguous K order (matches packed B): lanes 0-15 row M=lane hold K=k0..k0+15,
// lanes 16-31 row M=lane-16 hold K=k0+16..k0+31.
__device__ __forceinline__ v16h load_a_frag(const _Float16* tile, int ldk, int k0, int lane) {
    int row = lane & 15;
    int kb  = k0 + ((lane & 16) ? 16 : 0);
    const v8h* p0 = (const v8h*)(tile + row * ldk + kb);
    const v8h* p1 = (const v8h*)(tile + row * ldk + kb + 8);
    v8h lo = *p0, hi = *p1;
    v16h r;
#pragma unroll
    for (int i = 0; i < 8; ++i) { r[i] = lo[i]; r[8 + i] = hi[i]; }
    return r;
}

// ---------------- utility kernels ----------------

__global__ void fill_kernel(float* p, float v, int n) {
    int i = blockIdx.x * 256 + threadIdx.x;
    if (i < n) p[i] = v;
}

__global__ void degree_kernel(const int* dst, float* deg, int E) {
    int e = blockIdx.x * 256 + threadIdx.x;
    if (e < E) atomicAdd(&deg[dst[e]], 1.0f);
}

__global__ void invdeg_kernel(float* d, int n) {
    int i = blockIdx.x * 256 + threadIdx.x;
    if (i < n) { float v = d[i]; d[i] = (v > 0.0f) ? (1.0f / v) : 0.0f; }
}

// agg[dst] += h[src]; one thread per (edge, 4-feature chunk):
// one b128 load of h, 4 L2-resident f32 atomics, index loads amortized 4x.
__global__ void scatter_kernel(const int* __restrict__ src, const int* __restrict__ dst,
                               const float* __restrict__ h, float* __restrict__ agg, int total) {
    int idx = blockIdx.x * 256 + threadIdx.x;
    if (idx >= total) return;
    int e = idx >> 4, f = (idx & 15) * 4;
    v4f v = *(const v4f*)(h + src[e] * HD + f);
    float* ap = agg + dst[e] * HD + f;
    atomicAdd(ap + 0, v[0]);
    atomicAdd(ap + 1, v[1]);
    atomicAdd(ap + 2, v[2]);
    atomicAdd(ap + 3, v[3]);
}

// ---------------- weight packing (fragment layout) ----------------
// B fragment (32x16 f16): lane<16 -> col N=lane, halves i=0..15 = K 0..15;
// lane>=16 -> col N=lane-16, halves = K 16..31.  Frag element index:
// frag*512 + lane*16 + i.

__global__ void pack_gate_weights(const float* __restrict__ wl, const float* __restrict__ wr,
                                  const float* __restrict__ gw, const float* __restrict__ bl,
                                  const float* __restrict__ gb,
                                  _Float16* __restrict__ wpack, float* __restrict__ biasc) {
    int idx = blockIdx.x * 256 + threadIdx.x;  // 4*4*6*512 = 49152 elements
    if (idx < 49152) {
        int i = idx & 15, lane = (idx >> 4) & 31, frag = idx >> 9;
        int kk = frag % 6, t = (frag / 6) & 3, g = frag / 24;
        int n = t * 16 + (lane & 15);
        int k = kk * 32 + 16 * (lane >> 4) + i;  // 0..191
        float v;
        if (k < 64)        v = wl[(g * 64 + k) * 64 + n];
        else if (k < 128)  { int kr = k - 64; v = wr[(g * 64 + kr) * 64 + n] + gw[(g * 128 + kr) * 64 + n]; }
        else               { int kr = k - 128; v = gw[(g * 128 + 64 + kr) * 64 + n]; }
        wpack[idx] = (_Float16)v;
    }
    if (idx < 256) biasc[idx] = bl[idx] + gb[idx];
}

__global__ void pack_ec_weights(const float* __restrict__ w1, const float* __restrict__ w2,
                                _Float16* __restrict__ w1p, _Float16* __restrict__ w2p) {
    int idx = blockIdx.x * 256 + threadIdx.x;
    if (idx < 10240) {  // ec_w1: K padded 140->160 (5 ktiles), 4 ntiles
        int i = idx & 15, lane = (idx >> 4) & 31, frag = idx >> 9;
        int kk = frag % 5, t = frag / 5;
        int n = t * 16 + (lane & 15);
        int k = kk * 32 + 16 * (lane >> 4) + i;
        w1p[idx] = (k < 140) ? (_Float16)w1[k * 64 + n] : (_Float16)0.0f;
    }
    if (idx < 2048) {   // ec_w2: K=64 (2 ktiles), Nout=32 (2 ntiles)
        int i = idx & 15, lane = (idx >> 4) & 31, frag = idx >> 9;
        int kk = frag & 1, t = frag >> 1;
        int n = t * 16 + (lane & 15);
        int k = kk * 32 + 16 * (lane >> 4) + i;
        w2p[idx] = (_Float16)w2[k * 32 + n];
    }
}

// ---------------- node encoder (+ state init) ----------------

__device__ __forceinline__ float block64_sum(float v, float* red, int t) {
    __syncthreads();
    red[t] = v;
    __syncthreads();
#pragma unroll
    for (int s = 32; s > 0; s >>= 1) {
        if (t < s) red[t] += red[t + s];
        __syncthreads();
    }
    float r = red[0];
    __syncthreads();
    return r;
}

__global__ __launch_bounds__(64) void encoder_kernel(
    const float* __restrict__ x, const float* __restrict__ w1, const float* __restrict__ b1,
    const float* __restrict__ g1, const float* __restrict__ bn1,
    const float* __restrict__ w2, const float* __restrict__ b2,
    const float* __restrict__ g2, const float* __restrict__ bn2,
    float* __restrict__ h, float* __restrict__ cell,
    _Float16* __restrict__ h16, _Float16* __restrict__ enc16) {
    int node = blockIdx.x, t = threadIdx.x;
    __shared__ float xr[9];
    __shared__ float l1[64];
    __shared__ float red[64];
    if (t < 9) xr[t] = x[node * 9 + t];
    __syncthreads();

    float a = b1[t];
#pragma unroll
    for (int i = 0; i < 9; ++i) a = fmaf(xr[i], w1[i * 64 + t], a);
    a = fmaxf(a, 0.0f);
    float mu = block64_sum(a, red, t) * (1.0f / 64.0f);
    float var = block64_sum((a - mu) * (a - mu), red, t) * (1.0f / 64.0f);
    float v1 = (a - mu) * rsqrtf(var + 1e-5f) * g1[t] + bn1[t];
    l1[t] = v1;
    __syncthreads();

    float a2 = b2[t];
#pragma unroll 8
    for (int i = 0; i < 64; ++i) a2 = fmaf(l1[i], w2[i * 64 + t], a2);
    a2 = fmaxf(a2, 0.0f);
    float mu2 = block64_sum(a2, red, t) * (1.0f / 64.0f);
    float var2 = block64_sum((a2 - mu2) * (a2 - mu2), red, t) * (1.0f / 64.0f);
    float enc = (a2 - mu2) * rsqrtf(var2 + 1e-5f) * g2[t] + bn2[t];

    int o = node * HD + t;
    h[o] = enc;
    cell[o] = 0.0f;
    h16[o] = (_Float16)enc;
    enc16[o] = (_Float16)enc;
}

// ---------------- fused gates GEMM + LSTM + LN (WMMA) ----------------
// One block = 16 nodes. u = [agg*inv_deg | h | enc] (16x192 f16 in LDS).
// 4 waves, wave g computes gate g: 64 outputs = 4 n-tiles, K=192 = 6 k-tiles.

__global__ __launch_bounds__(128) void gates_kernel(
    const float* __restrict__ agg, const float* __restrict__ inv_deg,
    _Float16* __restrict__ h16, const _Float16* __restrict__ enc16,
    const _Float16* __restrict__ wpack, const float* __restrict__ biasc,
    const float* __restrict__ blk_g, const float* __restrict__ blk_b,
    float* __restrict__ h, float* __restrict__ cell, int N) {
    __shared__ __align__(16) _Float16 u[16 * 192];
    __shared__ float gbuf[4][16][64];
    __shared__ float rsum[128], rsq[128];

    int tid = threadIdx.x;
    int nodeBase = blockIdx.x * 16;

    // Stage u tile into LDS in 8-half (16B) chunks: 16 rows x 24 chunks.
    for (int j = tid; j < 16 * 24; j += 128) {
        int row = j / 24, c = j % 24;
        int node = nodeBase + row; if (node >= N) node = N - 1;
        v8h v;
        if (c < 8) {
            float id = inv_deg[node];
            v4f a0 = *(const v4f*)(agg + node * HD + c * 8);
            v4f a1 = *(const v4f*)(agg + node * HD + c * 8 + 4);
#pragma unroll
            for (int i = 0; i < 4; ++i) { v[i] = (_Float16)(a0[i] * id); v[4 + i] = (_Float16)(a1[i] * id); }
        } else if (c < 16) {
            v = *(const v8h*)(h16 + node * HD + (c - 8) * 8);
        } else {
            v = *(const v8h*)(enc16 + node * HD + (c - 16) * 8);
        }
        *(v8h*)(u + row * 192 + c * 8) = v;
    }
    __syncthreads();

    int wave = tid >> 5, lane = tid & 31;
    int g = wave;
    v8f zero = {};
    v8f acc[4] = {zero, zero, zero, zero};
#pragma unroll
    for (int kk = 0; kk < 6; ++kk) {
        v16h a = load_a_frag(u, 192, kk * 32, lane);
#pragma unroll
        for (int t = 0; t < 4; ++t) {
            v16h b = ((const v16h*)wpack)[((g * 4 + t) * 6 + kk) * 32 + lane];
            acc[t] = __builtin_amdgcn_wmma_f32_16x16x32_f16(
                false, a, false, b, (short)0, acc[t], false, false);
        }
    }
    // D layout: vgpr r, lanes 0-15 -> M=r N=lane; lanes 16-31 -> M=8+r N=lane-16.
#pragma unroll
    for (int t = 0; t < 4; ++t) {
        int n = t * 16 + (lane & 15);
        float bia = biasc[g * 64 + n];
#pragma unroll
        for (int r = 0; r < 8; ++r) {
            int m = (lane < 16) ? r : (r + 8);
            gbuf[g][m][n] = acc[t][r] + bia;
        }
    }
    __syncthreads();

    // LSTM update + LayerNorm: thread -> node tid/8, 8 features.
    int m = tid >> 3;
    int f0 = (tid & 7) * 8;
    int node = nodeBase + m;
    bool ok = node < N;
    int nr = ok ? node : (N - 1);
    float y[8];
    float s = 0.0f, q = 0.0f;
#pragma unroll
    for (int j = 0; j < 8; ++j) {
        int f = f0 + j;
        float fg = sigmoidf_(gbuf[0][m][f]);
        float ig = sigmoidf_(gbuf[1][m][f]);
        float cg = tanhf(gbuf[2][m][f]);
        float og = sigmoidf_(gbuf[3][m][f]);
        float cn = fg * cell[nr * HD + f] + ig * cg;
        if (ok) cell[node * HD + f] = cn;
        float yy = og * tanhf(cn);
        y[j] = yy; s += yy; q += yy * yy;
    }
    rsum[tid] = s; rsq[tid] = q;
    __syncthreads();
#pragma unroll
    for (int st = 4; st > 0; st >>= 1) {
        if ((tid & 7) < st) { rsum[tid] += rsum[tid + st]; rsq[tid] += rsq[tid + st]; }
        __syncthreads();
    }
    float mu = rsum[tid & ~7] * (1.0f / 64.0f);
    float var = rsq[tid & ~7] * (1.0f / 64.0f) - mu * mu;
    float rstd = rsqrtf(var + 1e-5f);
#pragma unroll
    for (int j = 0; j < 8; ++j) {
        int f = f0 + j;
        float hn = (y[j] - mu) * rstd * blk_g[f] + blk_b[f];
        if (ok) { h[node * HD + f] = hn; h16[node * HD + f] = (_Float16)hn; }
    }
}

// ---------------- edge classifier (WMMA) ----------------
// One wave per 16-edge tile, 4 tiles per block. K1 padded 140->160.

__global__ __launch_bounds__(128) void edge_kernel(
    const int* __restrict__ src, const int* __restrict__ dst,
    const float* __restrict__ edge_attr, const _Float16* __restrict__ h16,
    const _Float16* __restrict__ w1p, const float* __restrict__ ec_b1,
    const float* __restrict__ ec_g, const float* __restrict__ ec_bn,
    const _Float16* __restrict__ w2p, const float* __restrict__ ec_b2,
    const float* __restrict__ ec_w3, const float* __restrict__ ec_b3,
    float* __restrict__ logits, int E, int TILES) {
    __shared__ __align__(16) _Float16 a1[4][16 * 160];
    __shared__ float zbuf[4][16][64];
    __shared__ __align__(16) _Float16 a2[4][16 * 64];
    __shared__ float z2[4][16][32];

    int wave = threadIdx.x >> 5, lane = threadIdx.x & 31;
    int otile = blockIdx.x * 4 + wave;
    int tile = (otile < TILES) ? otile : (TILES - 1);   // clamp: keep EXEC full
    int eBase = tile * 16;
    _Float16* A1 = a1[wave];

    // Stage edge tile in 8-half (16B) chunks: 16 rows x 20 chunks.
    for (int j = lane; j < 16 * 20; j += 32) {
        int row = j / 20, c = j % 20;
        int e = eBase + row; if (e >= E) e = E - 1;
        v8h v;
        if (c < 8) {
            v = *(const v8h*)(h16 + src[e] * HD + c * 8);
        } else if (c < 16) {
            v = *(const v8h*)(h16 + dst[e] * HD + (c - 8) * 8);
        } else if (c == 16) {
            v4f a0 = *(const v4f*)(edge_attr + e * 12);
            v4f a1v = *(const v4f*)(edge_attr + e * 12 + 4);
#pragma unroll
            for (int i = 0; i < 4; ++i) { v[i] = (_Float16)a0[i]; v[4 + i] = (_Float16)a1v[i]; }
        } else if (c == 17) {
            v4f a2v = *(const v4f*)(edge_attr + e * 12 + 8);
#pragma unroll
            for (int i = 0; i < 4; ++i) { v[i] = (_Float16)a2v[i]; v[4 + i] = (_Float16)0.0f; }
        } else {
#pragma unroll
            for (int i = 0; i < 8; ++i) v[i] = (_Float16)0.0f;
        }
        *(v8h*)(A1 + row * 160 + c * 8) = v;
    }
    __syncthreads();

    // Layer 1: [16x160] x [160x64]
    v8f zero = {};
    v8f acc[4] = {zero, zero, zero, zero};
#pragma unroll
    for (int kk = 0; kk < 5; ++kk) {
        v16h a = load_a_frag(A1, 160, kk * 32, lane);
#pragma unroll
        for (int t = 0; t < 4; ++t) {
            v16h b = ((const v16h*)w1p)[(t * 5 + kk) * 32 + lane];
            acc[t] = __builtin_amdgcn_wmma_f32_16x16x32_f16(
                false, a, false, b, (short)0, acc[t], false, false);
        }
    }
#pragma unroll
    for (int t = 0; t < 4; ++t) {
        int n = t * 16 + (lane & 15);
        float bia = ec_b1[n];
#pragma unroll
        for (int r = 0; r < 8; ++r) {
            int m = (lane < 16) ? r : (r + 8);
            zbuf[wave][m][n] = fmaxf(acc[t][r] + bia, 0.0f);
        }
    }
    __syncthreads();

    // LayerNorm per edge: 2 lanes per edge, 32 feats each, combine via shfl.
    int ee = lane & 15, hp = lane >> 4;
    float s = 0.0f, q = 0.0f;
#pragma unroll 8
    for (int j = 0; j < 32; ++j) {
        float v = zbuf[wave][ee][hp * 32 + j];
        s += v; q += v * v;
    }
    s += __shfl_xor(s, 16);
    q += __shfl_xor(q, 16);
    float mu = s * (1.0f / 64.0f);
    float var = q * (1.0f / 64.0f) - mu * mu;
    float rstd = rsqrtf(var + 1e-5f);
#pragma unroll 8
    for (int j = 0; j < 32; ++j) {
        int f = hp * 32 + j;
        float zn = (zbuf[wave][ee][f] - mu) * rstd * ec_g[f] + ec_bn[f];
        a2[wave][ee * 64 + f] = (_Float16)zn;
    }
    __syncthreads();

    // Layer 2: [16x64] x [64x32]
    v8f acc2[2] = {zero, zero};
#pragma unroll
    for (int kk = 0; kk < 2; ++kk) {
        v16h a = load_a_frag(a2[wave], 64, kk * 32, lane);
#pragma unroll
        for (int t = 0; t < 2; ++t) {
            v16h b = ((const v16h*)w2p)[(t * 2 + kk) * 32 + lane];
            acc2[t] = __builtin_amdgcn_wmma_f32_16x16x32_f16(
                false, a, false, b, (short)0, acc2[t], false, false);
        }
    }
#pragma unroll
    for (int t = 0; t < 2; ++t) {
        int n = t * 16 + (lane & 15);
        float bia = ec_b2[n];
#pragma unroll
        for (int r = 0; r < 8; ++r) {
            int m = (lane < 16) ? r : (r + 8);
            z2[wave][m][n] = fmaxf(acc2[t][r] + bia, 0.0f);
        }
    }
    __syncthreads();

    // Layer 3: 32 -> 1
    float p = 0.0f;
#pragma unroll
    for (int j = 0; j < 16; ++j)
        p = fmaf(z2[wave][ee][hp * 16 + j], ec_w3[hp * 16 + j], p);
    p += __shfl_xor(p, 16);
    int eo = otile * 16 + ee;
    if (otile < TILES && lane < 16 && eo < E)
        logits[eo] = p + ec_b3[0];
}

// ---------------- host launcher ----------------

extern "C" void kernel_launch(void* const* d_in, const int* in_sizes, int n_in,
                              void* d_out, int out_size, void* d_ws, size_t ws_size,
                              hipStream_t stream) {
    const float* x        = (const float*)d_in[0];
    const int*   ei       = (const int*)d_in[1];
    const float* edge_at  = (const float*)d_in[2];
    const float* enc_w1   = (const float*)d_in[3];
    const float* enc_b1   = (const float*)d_in[4];
    const float* enc_g1   = (const float*)d_in[5];
    const float* enc_bn1  = (const float*)d_in[6];
    const float* enc_w2   = (const float*)d_in[7];
    const float* enc_b2   = (const float*)d_in[8];
    const float* enc_g2   = (const float*)d_in[9];
    const float* enc_bn2  = (const float*)d_in[10];
    const float* sage_wl  = (const float*)d_in[11];
    const float* sage_bl  = (const float*)d_in[12];
    const float* sage_wr  = (const float*)d_in[13];
    const float* gate_w   = (const float*)d_in[14];
    const float* gate_b   = (const float*)d_in[15];
    const float* blk_g    = (const float*)d_in[16];
    const float* blk_b    = (const float*)d_in[17];
    const float* ec_w1    = (const float*)d_in[18];
    const float* ec_b1    = (const float*)d_in[19];
    const float* ec_g     = (const float*)d_in[20];
    const float* ec_bn    = (const float*)d_in[21];
    const float* ec_w2    = (const float*)d_in[22];
    const float* ec_b2    = (const float*)d_in[23];
    const float* ec_w3    = (const float*)d_in[24];
    const float* ec_b3    = (const float*)d_in[25];
    float* logits = (float*)d_out;

    const int N = in_sizes[0] / 9;
    const int E = in_sizes[2] / 12;
    const int* src = ei;
    const int* dst = ei + E;

    // Workspace carve-out (256B aligned).
    size_t off = 0;
    auto alloc = [&](size_t bytes) {
        void* p = (char*)d_ws + off;
        off += (bytes + 255) & ~(size_t)255;
        return p;
    };
    float*     h      = (float*)alloc((size_t)N * HD * 4);
    float*     cell   = (float*)alloc((size_t)N * HD * 4);
    float*     agg    = (float*)alloc((size_t)N * HD * 4);
    _Float16*  h16    = (_Float16*)alloc((size_t)N * HD * 2);
    _Float16*  enc16  = (_Float16*)alloc((size_t)N * HD * 2);
    float*     invdeg = (float*)alloc((size_t)N * 4);
    _Float16*  wpack  = (_Float16*)alloc((size_t)49152 * 2);
    float*     biasc  = (float*)alloc((size_t)256 * 4);
    _Float16*  w1p    = (_Float16*)alloc((size_t)10240 * 2);
    _Float16*  w2p    = (_Float16*)alloc((size_t)2048 * 2);
    (void)ws_size; (void)n_in; (void)out_size;

    // Encoder + state init; weight packing; degree normalizer.
    encoder_kernel<<<N, 64, 0, stream>>>(x, enc_w1, enc_b1, enc_g1, enc_bn1,
                                         enc_w2, enc_b2, enc_g2, enc_bn2,
                                         h, cell, h16, enc16);
    pack_gate_weights<<<(49152 + 255) / 256, 256, 0, stream>>>(
        sage_wl, sage_wr, gate_w, sage_bl, gate_b, wpack, biasc);
    pack_ec_weights<<<(10240 + 255) / 256, 256, 0, stream>>>(ec_w1, ec_w2, w1p, w2p);
    fill_kernel<<<(N + 255) / 256, 256, 0, stream>>>(invdeg, 0.0f, N);
    degree_kernel<<<(E + 255) / 256, 256, 0, stream>>>(dst, invdeg, E);
    invdeg_kernel<<<(N + 255) / 256, 256, 0, stream>>>(invdeg, N);

    const int nodeTiles = (N + 15) / 16;
    const int nhFill = N * HD;
    const int scatterTotal = E * 16;   // (edge, float4-chunk) pairs
    for (int step = 0; step < 8; ++step) {
        fill_kernel<<<(nhFill + 255) / 256, 256, 0, stream>>>(agg, 0.0f, nhFill);
        scatter_kernel<<<(scatterTotal + 255) / 256, 256, 0, stream>>>(src, dst, h, agg, scatterTotal);
        gates_kernel<<<nodeTiles, 128, 0, stream>>>(agg, invdeg, h16, enc16,
                                                    wpack, biasc, blk_g, blk_b,
                                                    h, cell, N);
    }

    const int edgeTiles = (E + 15) / 16;
    edge_kernel<<<(edgeTiles + 3) / 4, 128, 0, stream>>>(
        src, dst, edge_at, h16, w1p, ec_b1, ec_g, ec_bn,
        w2p, ec_b2, ec_w3, ec_b3, logits, E, edgeTiles);
}
